// Self_Attention_38242388803969
// MI455X (gfx1250) — matrix operand order repaired and checked by reading, not compiled, not run on hardware
//
#include <hip/hip_runtime.h>

typedef __attribute__((ext_vector_type(16))) _Float16 v16h;
typedef __attribute__((ext_vector_type(8)))  float    v8f;
typedef __attribute__((ext_vector_type(8)))  _Float16 h8;
typedef __attribute__((ext_vector_type(4)))  int      v4i;

#define B_   8
#define C_   256
#define N_   4096
#define DK_  64
#define BM_  64
#define BN_  64
#define OD_  384              // 64 q + 64 k + 256 v output dims
#define SCALE_INV 0.125f      // 1/sqrt(64)

#define KLS  72               // padded K-tile LDS row stride (halves)
#define VTS  72               // padded V-tile LDS row stride (halves)
#define SLS  68               // padded S-tile LDS row stride (floats)

// ---------------- async global->LDS copy (gfx1250), with fallback ----------
#if defined(__has_builtin)
#if __has_builtin(__builtin_amdgcn_global_load_async_to_lds_b128) && \
    __has_builtin(__builtin_amdgcn_s_wait_asynccnt)
#define USE_ASYNC_LDS 1
#endif
#endif
#ifndef USE_ASYNC_LDS
#define USE_ASYNC_LDS 0
#endif

// copy 16 bytes (8 halves) global -> LDS
__device__ __forceinline__ void copy16(_Float16* lds, const _Float16* g) {
#if USE_ASYNC_LDS
  __builtin_amdgcn_global_load_async_to_lds_b128(
      (v4i*)(void*)const_cast<_Float16*>(g), (v4i*)(void*)lds, 0, 0);
#else
  *(h8*)lds = *(const h8*)g;
#endif
}
#if USE_ASYNC_LDS
#define ASYNC_WAIT(n) __builtin_amdgcn_s_wait_asynccnt(n)
#else
#define ASYNC_WAIT(n)
#endif

// Assemble a 16-half WMMA fragment from two contiguous 16-byte chunks.
__device__ __forceinline__ v16h make_frag(const _Float16* lo, const _Float16* hi) {
  h8 a = *(const h8*)lo;
  h8 b = *(const h8*)hi;
  v16h r;
#pragma unroll
  for (int i = 0; i < 8; i++) { r[i] = a[i]; r[i + 8] = b[i]; }
  return r;
}

__device__ __forceinline__ v8f wmma_f16(v16h a, v16h b, v8f c) {
  return __builtin_amdgcn_wmma_f32_16x16x32_f16(false, a, false, b, (short)0, c,
                                                false, false);
}

// ---------------------------------------------------------------------------
// Convert Wq/Wk/Wv -> f16 [384][256], biases -> f32 [384].
// ---------------------------------------------------------------------------
__global__ __launch_bounds__(256) void convert_w_kernel(
    const float* __restrict__ Wq, const float* __restrict__ bq,
    const float* __restrict__ Wk, const float* __restrict__ bk,
    const float* __restrict__ Wv, const float* __restrict__ bv,
    _Float16* __restrict__ Wh, float* __restrict__ bias_all)
{
  const int od = blockIdx.x;
  const int c  = threadIdx.x;
  float wv;
  if (od < 64)       wv = Wq[(size_t)od * C_ + c];
  else if (od < 128) wv = Wk[(size_t)(od - 64) * C_ + c];
  else               wv = Wv[(size_t)(od - 128) * C_ + c];
  Wh[(size_t)od * C_ + c] = (_Float16)wv;
  if (c == 0) {
    bias_all[od] = (od < 64) ? bq[od] : (od < 128) ? bk[od - 64] : bv[od - 128];
  }
}

// ---------------------------------------------------------------------------
// Transpose + convert: x [B][C][N] f32  ->  xh [B][N][C] f16.
// ---------------------------------------------------------------------------
__global__ __launch_bounds__(256) void transpose_x_kernel(
    const float* __restrict__ x, _Float16* __restrict__ xh)
{
  __shared__ float ts[64][65];                // padded to dodge bank conflicts
  const int blk = blockIdx.x;                 // b * 64 * 4 tiles
  const int b   = blk / (64 * 4);
  const int rem = blk % (64 * 4);
  const int nbase = (rem / 4) * 64;
  const int cbase = (rem % 4) * 64;
  const int t = threadIdx.x;
  {
    const int i  = t >> 2;                    // c index
    const int j0 = (t & 3) * 16;              // n index
    const float* xp = x + ((size_t)b * C_ + cbase + i) * N_ + nbase + j0;
#pragma unroll
    for (int j = 0; j < 16; j++) ts[i][j0 + j] = xp[j];
  }
  __syncthreads();
  {
    const int n  = t >> 2;
    const int c0 = (t & 3) * 16;
    _Float16* op = xh + ((size_t)b * N_ + nbase + n) * C_ + cbase + c0;
#pragma unroll
    for (int j = 0; j < 16; j++) op[j] = (_Float16)ts[c0 + j][n];
  }
}

// ---------------------------------------------------------------------------
// Fused Q/K/V projection GEMM via WMMA.
//   Y[pos, od] = sum_c xh[pos,c] * Wh[od,c] + bias[od],  od in [0,384)
//   od 0..63 -> Qb [b][n][64], 64..127 -> Kb [b][n][64],
//   128..383 -> VtG [b][n/64][c][key]  (V stored pre-transposed per key-tile)
// ---------------------------------------------------------------------------
__global__ __launch_bounds__(256) void proj_wmma_kernel(
    const _Float16* __restrict__ xh, const _Float16* __restrict__ Wh,
    const float* __restrict__ bias_all,
    _Float16* __restrict__ Qb, _Float16* __restrict__ Kb,
    _Float16* __restrict__ VtG)
{
  __shared__ _Float16 xt[64 * 256];           // 32 KB x tile [pos][c]
  const int blk   = blockIdx.x;
  const int b     = blk / (N_ / 64);
  const int pbase = (blk % (N_ / 64)) * 64;
  const int t     = threadIdx.x;
  const int w     = t >> 5;
  const int l     = t & 31;
  const int half  = l >> 4;
  const int lnid  = l & 15;

  {                                           // stage x tile (contiguous 32KB)
    const _Float16* src = xh + ((size_t)b * N_ + pbase) * C_ + (size_t)t * 64;
    _Float16* dst = &xt[t * 64];
#pragma unroll
    for (int j = 0; j < 8; j++) copy16(dst + j * 8, src + j * 8);
  }
  ASYNC_WAIT(0);
  __syncthreads();

  v8f acc[4][3];
#pragma unroll
  for (int rt = 0; rt < 4; rt++)
#pragma unroll
    for (int ct = 0; ct < 3; ct++) acc[rt][ct] = (v8f)0.f;

  for (int s = 0; s < 8; s++) {               // K chunks of 32 over C=256
    v16h bf[3];
#pragma unroll
    for (int ct = 0; ct < 3; ct++) {          // B = W^T: lane -> od column
      const int od = (3 * w + ct) * 16 + lnid;
      const _Float16* wp = Wh + (size_t)od * C_ + 32 * s + 8 * half;
      bf[ct] = make_frag(wp, wp + 16);
    }
#pragma unroll
    for (int rt = 0; rt < 4; rt++) {
      const int row = rt * 16 + lnid;
      v16h af = make_frag(&xt[row * C_ + 32 * s + 8 * half],
                          &xt[row * C_ + 32 * s + 16 + 8 * half]);
#pragma unroll
      for (int ct = 0; ct < 3; ct++)
        acc[rt][ct] = wmma_f16(af, bf[ct], acc[rt][ct]);
    }
  }

#pragma unroll
  for (int ct = 0; ct < 3; ct++) {
    const int ot = 3 * w + ct;
    const int od = ot * 16 + lnid;
    const float bia = bias_all[od];
    if (ot < 8) {                             // Q (ot<4) or K tiles
      _Float16* dstb = (ot < 4) ? Qb : Kb;
      const int odl  = (ot < 4) ? od : od - 64;
#pragma unroll
      for (int rt = 0; rt < 4; rt++)
#pragma unroll
        for (int r = 0; r < 8; r++) {
          const int pos = pbase + rt * 16 + r + 8 * half;
          dstb[((size_t)b * N_ + pos) * DK_ + odl] =
              (_Float16)(acc[rt][ct][r] + bia);
        }
    } else {                                  // V: store transposed, packed h8
      const int c = od - 128;
#pragma unroll
      for (int rt = 0; rt < 4; rt++) {
        h8 pk;
#pragma unroll
        for (int r = 0; r < 8; r++) pk[r] = (_Float16)(acc[rt][ct][r] + bia);
        _Float16* vp = VtG +
            (((size_t)(b * (N_ / 64) + (pbase >> 6)) * C_ + c) * 64) +
            rt * 16 + 8 * half;
        *(h8*)vp = pk;
      }
    }
  }
}

// ---------------------------------------------------------------------------
// Issue async copies of one (K,V) key-tile into padded LDS buffers.
// Per thread: 2 (K) + 8 (V) = 10 async b128 copies.
// ---------------------------------------------------------------------------
__device__ __forceinline__ void issue_tile(
    const _Float16* __restrict__ Kb, const _Float16* __restrict__ VtG,
    int b, int kb, int t, _Float16* Klb, _Float16* Vtb)
{
  {                                   // K tile: 4 threads/row, 2 x 16B each
    const int key = t >> 2;
    const int d0  = (t & 3) * 16;
    const _Float16* ksrc = Kb + ((size_t)b * N_ + kb + key) * DK_ + d0;
    copy16(&Klb[key * KLS + d0], ksrc);
    copy16(&Klb[key * KLS + d0 + 8], ksrc + 8);
  }
  {                                   // V tile: 1 thread/row (c=t), 8 x 16B
    const _Float16* vsrc = VtG +
        ((size_t)(b * (N_ / 64) + (kb >> 6)) * C_ + t) * 64;
    _Float16* vdst = &Vtb[t * VTS];
#pragma unroll
    for (int j = 0; j < 8; j++) copy16(vdst + j * 8, vsrc + j * 8);
  }
}

// ---------------------------------------------------------------------------
// Flash attention. One block per (batch, 64-query tile). 8 waves.
//   wave w: S-phase tiles (qt = w&3, kt = 2*(w>>2)+{0,1}); PV-phase owns
//   output channels [32w, 32w+32). K/V tiles double-buffered via async LDS.
// ---------------------------------------------------------------------------
__global__ __launch_bounds__(256) void flash_kernel(
    const _Float16* __restrict__ Qb, const _Float16* __restrict__ Kb,
    const _Float16* __restrict__ VtG, const float* __restrict__ x,
    const float* __restrict__ gamma, float* __restrict__ out)
{
  __shared__ _Float16 Kl[2][64 * KLS];   // K tiles [key][d] (padded)
  __shared__ _Float16 Vt[2][256 * VTS];  // V tiles [c][key] (padded)
  __shared__ float    Sl[64 * SLS];      // f32 scores; f16 P aliases row starts
  __shared__ float    alpha_s[64];
  __shared__ float    mrow_s[64];
  __shared__ float    l_s[64];
  __shared__ float    pm[4 * 64];        // per-segment partial max / sum

  const int blk   = blockIdx.x;
  const int b     = blk / (N_ / BM_);
  const int qbase = (blk % (N_ / BM_)) * BM_;
  const int t     = threadIdx.x;
  const int w     = t >> 5;
  const int l     = t & 31;
  const int half  = l >> 4;
  const int lnid  = l & 15;
  const int row64 = t & 63;              // softmax: row owned segment-wise
  const int seg   = t >> 6;              // 4 segments of 16 keys

  // ---- stage Q tile into Kl[0], gather Q A-frags ----
  {
    const int q  = t >> 2;
    const int d0 = (t & 3) * 16;
    const _Float16* qp = Qb + ((size_t)b * N_ + qbase + q) * DK_ + d0;
    copy16(&Kl[0][q * KLS + d0], qp);
    copy16(&Kl[0][q * KLS + d0 + 8], qp + 8);
  }
  ASYNC_WAIT(0);
  __syncthreads();
  const int qt_s = w & 3;
  const int kt0  = (w >> 2) * 2;
  v16h qf[2];
#pragma unroll
  for (int s = 0; s < 2; s++) {
    const int row = qt_s * 16 + lnid;
    qf[s] = make_frag(&Kl[0][row * KLS + 32 * s + 8 * half],
                      &Kl[0][row * KLS + 32 * s + 16 + 8 * half]);
  }
  __syncthreads();                       // all Q reads done before tile0 lands

  v8f acc[8];
#pragma unroll
  for (int i = 0; i < 8; i++) acc[i] = (v8f)0.f;
  float m_r = -__builtin_inff();
  float l_r = 0.f;

  issue_tile(Kb, VtG, b, 0, t, Kl[0], Vt[0]);   // prefetch tile 0

  for (int kb = 0; kb < N_; kb += BN_) {
    const int cb = (kb >> 6) & 1;
    if (kb + BN_ < N_) {                 // prefetch next tile, then wait current
      issue_tile(Kb, VtG, b, kb + BN_, t, Kl[cb ^ 1], Vt[cb ^ 1]);
      ASYNC_WAIT(10);                    // in-order: current tile's 10 retired
    } else {
      ASYNC_WAIT(0);
    }
    __syncthreads();
    const _Float16* Klc = Kl[cb];
    const _Float16* Vtc = Vt[cb];

    // -- S = Q·K^T: each wave 2 tiles, K-dim 64 = 2 WMMA steps --
#pragma unroll
    for (int kk = 0; kk < 2; kk++) {
      const int kt = kt0 + kk;
      v8f s_acc = (v8f)0.f;
#pragma unroll
      for (int s = 0; s < 2; s++) {
        const int key = kt * 16 + lnid;  // B: lane -> key col, elem i -> d
        v16h bfr = make_frag(&Klc[key * KLS + 32 * s + 8 * half],
                             &Klc[key * KLS + 32 * s + 16 + 8 * half]);
        s_acc = wmma_f16(qf[s], bfr, s_acc);
      }
#pragma unroll
      for (int r = 0; r < 8; r++)
        Sl[(qt_s * 16 + r + 8 * half) * SLS + kt * 16 + lnid] = s_acc[r];
    }
    __syncthreads();

    // -- online softmax, 4 threads per row --
    {
      float pmax = -__builtin_inff();
#pragma unroll
      for (int j = 0; j < 16; j++)
        pmax = fmaxf(pmax, Sl[row64 * SLS + seg * 16 + j] * SCALE_INV);
      pm[seg * 64 + row64] = pmax;
    }
    __syncthreads();
    if (t < 64) {
      const float mnew = fmaxf(m_r,
          fmaxf(fmaxf(pm[t], pm[64 + t]), fmaxf(pm[128 + t], pm[192 + t])));
      alpha_s[t] = __expf(m_r - mnew);   // 0 on first tile
      mrow_s[t]  = mnew;
      m_r = mnew;
    }
    __syncthreads();
    {
      float sv[16];
#pragma unroll
      for (int j = 0; j < 16; j++)
        sv[j] = Sl[row64 * SLS + seg * 16 + j] * SCALE_INV;
      const float mn = mrow_s[row64];
      __syncthreads();                   // all f32 reads done before f16 writes
      _Float16* prow = (_Float16*)&Sl[row64 * SLS];
      float psum = 0.f;
#pragma unroll
      for (int j = 0; j < 16; j++) {
        const float p = __expf(sv[j] - mn);
        psum += p;
        prow[seg * 16 + j] = (_Float16)p;
      }
      pm[seg * 64 + row64] = psum;
    }
    __syncthreads();
    if (t < 64)
      l_r = l_r * alpha_s[t] + (pm[t] + pm[64 + t]) + (pm[128 + t] + pm[192 + t]);

    // -- rescale O accumulators, then O += P · V --
#pragma unroll
    for (int qt = 0; qt < 4; qt++) {
      float al[8];
#pragma unroll
      for (int r = 0; r < 8; r++) al[r] = alpha_s[qt * 16 + r + 8 * half];
#pragma unroll
      for (int ct = 0; ct < 2; ct++)
#pragma unroll
        for (int r = 0; r < 8; r++) acc[qt * 2 + ct][r] *= al[r];
    }
    v16h vf[2][2];
#pragma unroll
    for (int ct = 0; ct < 2; ct++)
#pragma unroll
      for (int s = 0; s < 2; s++) {
        const int c = w * 32 + ct * 16 + lnid;
        vf[ct][s] = make_frag(&Vtc[c * VTS + 32 * s + 8 * half],
                              &Vtc[c * VTS + 32 * s + 16 + 8 * half]);
      }
#pragma unroll
    for (int qt = 0; qt < 4; qt++) {
#pragma unroll
      for (int s = 0; s < 2; s++) {
        const _Float16* prow = (const _Float16*)&Sl[(qt * 16 + lnid) * SLS];
        v16h pf = make_frag(prow + 32 * s + 8 * half,
                            prow + 32 * s + 16 + 8 * half);
#pragma unroll
        for (int ct = 0; ct < 2; ct++)
          acc[qt * 2 + ct] = wmma_f16(pf, vf[ct][s], acc[qt * 2 + ct]);
      }
    }
    __syncthreads();                     // buffer cb free for prefetch reuse
  }

  // ---- epilogue: out[b,c,m] = x[b,c,m] + gamma * O[m,c] / l[m] ----
  if (t < 64) l_s[t] = l_r;
  __syncthreads();
  const float gm = gamma[0];
  float* Ost = Sl;                       // reuse as flat [16 q][256 c] staging
#pragma unroll 1
  for (int qt = 0; qt < 4; qt++) {
#pragma unroll
    for (int ct = 0; ct < 2; ct++)
#pragma unroll
      for (int r = 0; r < 8; r++) {
        const int qrow = r + 8 * half;
        const int c    = w * 32 + ct * 16 + lnid;
        Ost[qrow * 256 + c] = acc[qt * 2 + ct][r] / l_s[qt * 16 + qrow];
      }
    __syncthreads();
    {
      const size_t base = ((size_t)b * C_ + t) * N_ + qbase + qt * 16;
#pragma unroll
      for (int q = 0; q < 16; q++)
        out[base + q] = x[base + q] + gm * Ost[q * 256 + t];
    }
    __syncthreads();
  }
}

// ---------------------------------------------------------------------------
extern "C" void kernel_launch(void* const* d_in, const int* in_sizes, int n_in,
                              void* d_out, int out_size, void* d_ws, size_t ws_size,
                              hipStream_t stream) {
  const float* x     = (const float*)d_in[0];
  const float* Wq    = (const float*)d_in[1];
  const float* bq    = (const float*)d_in[2];
  const float* Wk    = (const float*)d_in[3];
  const float* bk    = (const float*)d_in[4];
  const float* Wv    = (const float*)d_in[5];
  const float* bv    = (const float*)d_in[6];
  const float* gamma = (const float*)d_in[7];
  float* out = (float*)d_out;

  _Float16* Qb  = (_Float16*)d_ws;                          //  4 MB
  _Float16* Kb  = Qb  + (size_t)B_ * N_ * DK_;              //  4 MB
  _Float16* VtG = Kb  + (size_t)B_ * N_ * DK_;              // 16 MB (transposed V)
  _Float16* xh  = VtG + (size_t)B_ * N_ * C_;               // 16 MB
  _Float16* Wh  = xh  + (size_t)B_ * N_ * C_;               // 192 KB
  float*    bias_all = (float*)(Wh + (size_t)OD_ * C_);     // 1.5 KB

  convert_w_kernel <<<OD_, 256, 0, stream>>>(Wq, bq, Wk, bk, Wv, bv, Wh, bias_all);
  transpose_x_kernel<<<B_ * 64 * 4, 256, 0, stream>>>(x, xh);
  proj_wmma_kernel <<<B_ * (N_ / 64), 256, 0, stream>>>(xh, Wh, bias_all, Qb, Kb, VtG);
  flash_kernel     <<<B_ * (N_ / BM_), 256, 0, stream>>>(Qb, Kb, VtG, x, gamma, out);
}
